// WaveAttention_79147657331172
// MI455X (gfx1250) — matrix operand order repaired and checked by reading, not compiled, not run on hardware
//
#include <hip/hip_runtime.h>
#include <hip/hip_bf16.h>
#include <stdint.h>

typedef __bf16 bf16_t;
typedef __attribute__((ext_vector_type(16))) __bf16 v16bf;
typedef __attribute__((ext_vector_type(8)))  float  v8f;
typedef __attribute__((ext_vector_type(4)))  unsigned int v4u;
typedef __attribute__((ext_vector_type(8)))  int v8i;
typedef __attribute__((ext_vector_type(4)))  int v4i;

// ---- helpers -------------------------------------------------------------

__device__ __forceinline__ bf16_t f2bf(float f) { return (bf16_t)f; }  // native cvt

union FragBF { v16bf v; uint32_t u[8]; };

// A-matrix 16x32 bf16 fragment (ISA 7.12.2): lane<16 holds K={0..7,16..23},
// lane>=16 holds K={8..15,24..31}; rowp = 16 dword-pairs of one M-row.
__device__ __forceinline__ v16bf frag_a_k32(const uint32_t* rowp, int hi) {
  FragBF f;
  const int b0 = hi * 4;
#pragma unroll
  for (int j = 0; j < 4; ++j) f.u[j]     = rowp[b0 + j];
#pragma unroll
  for (int j = 0; j < 4; ++j) f.u[4 + j] = rowp[b0 + 8 + j];
  return f.v;
}

// B-matrix 32x16 bf16 fragment: lane<16 holds K={0..7,16..23}, lane>=16 K={8..15,24..31}.
__device__ __forceinline__ v16bf frag_b_k32(const uint32_t* colp, int hi) {
  FragBF f;
  const int b0 = hi * 4;
#pragma unroll
  for (int j = 0; j < 8; ++j) f.u[j] = colp[b0 + (j & 3) + ((j >> 2) << 3)];
  return f.v;
}

// LDS byte address for TDM descriptors: ISA §10.2 — LDS aperture maps
// the low 32 flat-address bits to the LDS offset, so the low dword of the generic pointer works.
__device__ __forceinline__ uint32_t lds_addr_u32(const void* p) {
  return (uint32_t)(uintptr_t)p;
}

// Issue a TDM 2D-tile load (bf16 elements) global -> LDS.
// D# per CDNA5 ISA: group0 = {count|flags, lds_addr, gaddr lo, gaddr hi|type=2}
// group1 packs data_size / tensor dims / tile dims / dim0 stride.
__device__ __forceinline__ void tdm_load_2d_bf16(uint32_t lds_off, const void* gptr,
                                                 uint32_t tile_d0, uint32_t tile_d1,
                                                 uint32_t stride_elems) {
  uint64_t ga = (uint64_t)(uintptr_t)gptr;
  v4u g0;
  g0.x = 1u;                                   // count=1 (valid), user mode
  g0.y = lds_off;                              // lds_addr (bytes)
  g0.z = (uint32_t)ga;                         // global_addr[31:0]
  g0.w = (uint32_t)(ga >> 32) | (2u << 30);    // global_addr[56:32] | type=2
  v8i g1;
  g1[0] = (int)(1u << 16);                                    // data_size=2B
  g1[1] = (int)((tile_d0 & 0xffffu) << 16);                   // tensor_dim0 lo
  g1[2] = (int)((tile_d0 >> 16) | ((tile_d1 & 0xffffu) << 16)); // d0 hi|d1 lo
  g1[3] = (int)((tile_d1 >> 16) | (tile_d0 << 16));           // d1 hi|tile_dim0
  g1[4] = (int)(tile_d1);                                     // tile_dim1, dim2=0
  g1[5] = (int)stride_elems;                                  // dim0_stride lo32
  g1[6] = 0;                                                  // stride hi | d1s lo
  g1[7] = 0;
  v4i g2 = {0, 0, 0, 0};
  v4i g3 = {0, 0, 0, 0};
#if defined(__clang_major__) && __clang_major__ >= 23
  v8i g4 = {0, 0, 0, 0, 0, 0, 0, 0};
  __builtin_amdgcn_tensor_load_to_lds(g0, g1, g2, g3, g4, 0);
#else
  __builtin_amdgcn_tensor_load_to_lds(g0, g1, g2, g3, 0);
#endif
}

// ---- kernel 0: one-pass f32 -> bf16 convert ------------------------------

__global__ __launch_bounds__(256) void cvt_f32_bf16(const float* __restrict__ src,
                                                    bf16_t* __restrict__ dst) {
  size_t i = ((size_t)blockIdx.x * 256 + threadIdx.x) * 4;
  float4 f = *(const float4*)(src + i);
  union { bf16_t b[4]; uint2 u; } o;
  o.b[0] = f2bf(f.x); o.b[1] = f2bf(f.y); o.b[2] = f2bf(f.z); o.b[3] = f2bf(f.w);
  *(uint2*)(dst + i) = o.u;
}

// ---- kernel 1: QKV GEMM, TDM double-buffered -----------------------------
// C = x[200704,256] @ w[768,256]^T + b ; block tile 128(M) x 64(N), K-step 32.
// Wave 0 drives the Tensor Data Mover; WMMAs overlap the in-flight DMA.

#define K1_BM 128
#define K1_BN 64
#define K1_BK 32
#define K1_ITERS 8

__global__ __launch_bounds__(256) void qkv_gemm_bf16(
    const bf16_t* __restrict__ xbf, const bf16_t* __restrict__ wbf,
    const float* __restrict__ bias, bf16_t* __restrict__ qkv)
{
  __shared__ bf16_t As[2][K1_BM][K1_BK];   // 2 x 8 KB
  __shared__ bf16_t Bs[2][K1_BN][K1_BK];   // 2 x 4 KB

  const int tid  = threadIdx.x;
  const int lane = tid & 31;
  const int wave = tid >> 5;
  const int hi   = lane >> 4;
  const int l16  = lane & 15;
  const size_t m0 = (size_t)blockIdx.x * K1_BM;
  const int    n0 = blockIdx.y * K1_BN;

  v8f acc[4] = {};

  if (wave == 0) {   // prologue: tile 0 -> buffer 0
    tdm_load_2d_bf16(lds_addr_u32(&As[0][0][0]), xbf + m0 * 256,
                     K1_BK, K1_BM, 256);
    tdm_load_2d_bf16(lds_addr_u32(&Bs[0][0][0]), wbf + (size_t)n0 * 256,
                     K1_BK, K1_BN, 256);
  }

  for (int i = 0; i < K1_ITERS; ++i) {
    const int buf = i & 1;
    if (wave == 0) {
      if (i + 1 < K1_ITERS) {     // issue next pair, then wait for previous pair
        const int k0n = (i + 1) * K1_BK;
        tdm_load_2d_bf16(lds_addr_u32(&As[buf ^ 1][0][0]),
                         xbf + m0 * 256 + k0n, K1_BK, K1_BM, 256);
        tdm_load_2d_bf16(lds_addr_u32(&Bs[buf ^ 1][0][0]),
                         wbf + (size_t)n0 * 256 + k0n, K1_BK, K1_BN, 256);
        __builtin_amdgcn_s_wait_tensorcnt(2);   // in-order: first pair retired
      } else {
        __builtin_amdgcn_s_wait_tensorcnt(0);
      }
    }
    __syncthreads();              // buf tile resident for everyone

    v16bf av = frag_a_k32((const uint32_t*)&As[buf][wave * 16 + l16][0], hi);
#pragma unroll
    for (int t = 0; t < 4; ++t) {
      v16bf bv = frag_b_k32((const uint32_t*)&Bs[buf][t * 16 + l16][0], hi);
      acc[t] = __builtin_amdgcn_wmma_f32_16x16x32_bf16(
          false, av, false, bv, (short)0, acc[t], false, false);
    }
    __syncthreads();              // tile reads done before buf is re-filled
  }

  // C/D layout: VGPR v -> M = v + (lane>=16 ? 8:0), N = lane&15
  const size_t mrow = m0 + (size_t)wave * 16 + hi * 8;
#pragma unroll
  for (int t = 0; t < 4; ++t) {
    int   n  = n0 + t * 16 + l16;
    float bv = bias[n];
#pragma unroll
    for (int v = 0; v < 8; ++v)
      qkv[(mrow + v) * 768 + n] = f2bf(acc[t][v] + bv);
  }
}

// ---- kernel 2: fused shifted-window attention ----------------------------
// 1 block = 1 window (4096 blocks), 8 waves = 8 heads. Dynamic LDS:
//  Q  bf16 [8][64][32]  @      0  (32 KB)   (tokens 49..63 zero padded)
//  K  bf16 [8][64][32]  @  32768  (32 KB)
//  Vt bf16 [8][32][64]  @  65536  (32 KB)   (transposed: K-pairs contiguous)
//  S  f32  [8][64][64]  @  98304  (128 KB)  (P bf16 overwrites each row in place)
//  Rs f32  [8][64]      @ 229376  (2 KB)    row exp-sums

__global__ __launch_bounds__(256) void win_attn(
    const bf16_t* __restrict__ qkv, const float* __restrict__ mask,
    const float* __restrict__ rel_table, float* __restrict__ out)
{
  extern __shared__ char smem[];
  char*  Qb = smem;
  char*  Kb = smem + 32768;
  char*  Vb = smem + 65536;
  float* Sf = (float*)(smem + 98304);
  float* Rs = (float*)(smem + 229376);

  const int tid  = threadIdx.x;
  const int lane = tid & 31;
  const int h    = tid >> 5;          // wave == head
  const int hi   = lane >> 4;
  const int l16  = lane & 15;

  const int win  = blockIdx.x & 1023;
  const int b    = blockIdx.x >> 10;
  const int wh7  = (win >> 5) * 7;
  const int ww7  = (win & 31) * 7;
  const size_t tokbase = (size_t)b * 50176;

  const uint16_t* qkvu = (const uint16_t*)qkv;

  // ---- stage Q,K (b64 chunks, shift-remapped gather, zero pad t>=49) ----
  for (int i = tid; i < 2 * 8 * 64 * 8; i += 256) {
    int mat = i >> 12;
    int rem = i & 4095;
    int hh  = rem >> 9;
    int t   = (rem >> 3) & 63;
    int dc  = (rem & 7) << 2;
    uint2 val; val.x = 0u; val.y = 0u;
    if (t < 49) {
      int r = t / 7, c = t % 7;
      int gr = wh7 + r + 3; if (gr >= 224) gr -= 224;
      int gc = ww7 + c + 3; if (gc >= 224) gc -= 224;
      size_t g = tokbase + (size_t)gr * 224 + gc;
      val = *(const uint2*)(qkvu + g * 768 + mat * 256 + hh * 32 + dc);
    }
    *(uint2*)((mat ? Kb : Qb) + (size_t)(((hh * 64 + t) * 32 + dc) * 2)) = val;
  }
  // ---- stage V transposed ----
  for (int i = tid; i < 8 * 64 * 32; i += 256) {
    int hh = i >> 11;
    int t  = (i >> 5) & 63;
    int d  = i & 31;
    uint16_t val = 0;
    if (t < 49) {
      int r = t / 7, c = t % 7;
      int gr = wh7 + r + 3; if (gr >= 224) gr -= 224;
      int gc = ww7 + c + 3; if (gc >= 224) gc -= 224;
      size_t g = tokbase + (size_t)gr * 224 + gc;
      val = qkvu[g * 768 + 512 + hh * 32 + d];
    }
    *(uint16_t*)(Vb + (size_t)(((hh * 32 + d) * 64 + t) * 2)) = val;
  }
  __syncthreads();

  // ---- S = (Q K^T) * scale : 4x4 tiles, one WMMA each (K = hd = 32) ----
  const float scale = 0.17677669529663687f;  // 32^-0.5
  v8f accS[4][4] = {};
  v16bf aq[4];
#pragma unroll
  for (int mt = 0; mt < 4; ++mt)
    aq[mt] = frag_a_k32(
        (const uint32_t*)(Qb + (size_t)((h * 64 + mt * 16 + l16) * 64)), hi);
#pragma unroll
  for (int nt = 0; nt < 4; ++nt) {
    v16bf bk = frag_b_k32(
        (const uint32_t*)(Kb + (size_t)((h * 64 + nt * 16 + l16) * 64)), hi);
#pragma unroll
    for (int mt = 0; mt < 4; ++mt)
      accS[mt][nt] = __builtin_amdgcn_wmma_f32_16x16x32_bf16(
          false, aq[mt], false, bk, (short)0, accS[mt][nt], false, false);
  }
#pragma unroll
  for (int mt = 0; mt < 4; ++mt)
#pragma unroll
    for (int nt = 0; nt < 4; ++nt) {
      int qb = mt * 16 + hi * 8;
      int n  = nt * 16 + l16;
#pragma unroll
      for (int v = 0; v < 8; ++v)
        Sf[(h * 64 + qb + v) * 64 + n] = accS[mt][nt][v] * scale;
    }
  __syncthreads();

  // ---- softmax: + rel-pos bias (computed in kernel) + mask --------------
  for (int rt = tid; rt < 512; rt += 256) {
    int hh = rt >> 6;
    int q  = rt & 63;
    float* Srow = Sf + (hh * 64 + q) * 64;
    float rsum = 1.0f;
    if (q < 49) {
      int qr = q / 7, qc = q % 7;
      const float* mrow = mask + ((size_t)win * 49 + q) * 49;
      float mx = -3.0e30f;
      for (int n = 0; n < 49; ++n) {
        int kr = n / 7, kc = n % 7;
        int ridx = (qr - kr + 6) * 13 + (qc - kc + 6);
        float vv = Srow[n] + rel_table[ridx * 8 + hh] + mrow[n];
        Srow[n] = vv;
        mx = fmaxf(mx, vv);
      }
      float s = 0.f;
      bf16_t* Prow = (bf16_t*)Srow;   // bf16 writes trail f32 reads: safe
      for (int n = 0; n < 64; ++n) {
        float e = 0.f;
        if (n < 49) { e = __expf(Srow[n] - mx); s += e; }
        Prow[n] = f2bf(e);
      }
      rsum = s;
    } else {
      uint32_t* Prow = (uint32_t*)Srow;
      for (int j = 0; j < 32; ++j) Prow[j] = 0u;  // zero padded P rows
    }
    Rs[hh * 64 + q] = rsum;
  }
  __syncthreads();

  // ---- O = P V : 4x2 tiles, K=64 in two 32-steps ------------------------
  v8f accO[4][2] = {};
  v16bf bv[2][2];
#pragma unroll
  for (int nt = 0; nt < 2; ++nt) {
    const uint32_t* colp = (const uint32_t*)(
        Vb + (size_t)(((h * 32 + nt * 16 + l16) * 64) * 2));
#pragma unroll
    for (int kk = 0; kk < 2; ++kk)
      bv[nt][kk] = frag_b_k32(colp + kk * 16, hi);
  }
#pragma unroll
  for (int mt = 0; mt < 4; ++mt) {
    const uint32_t* rowp =
        (const uint32_t*)(Sf + (h * 64 + mt * 16 + l16) * 64);  // P bf16 row
#pragma unroll
    for (int kk = 0; kk < 2; ++kk) {
      v16bf pa = frag_a_k32(rowp + kk * 16, hi);
#pragma unroll
      for (int nt = 0; nt < 2; ++nt)
        accO[mt][nt] = __builtin_amdgcn_wmma_f32_16x16x32_bf16(
            false, pa, false, bv[nt][kk], (short)0, accO[mt][nt], false, false);
    }
  }

  // ---- epilogue: divide by rowsum, scatter through shift remap ----------
#pragma unroll
  for (int mt = 0; mt < 4; ++mt) {
    int qb = mt * 16 + hi * 8;
#pragma unroll
    for (int v = 0; v < 8; ++v) {
      int q = qb + v;
      if (q < 49) {
        float inv = 1.0f / Rs[h * 64 + q];
        int r = q / 7, c = q % 7;
        int gr = wh7 + r + 3; if (gr >= 224) gr -= 224;
        int gc = ww7 + c + 3; if (gc >= 224) gc -= 224;
        size_t g = tokbase + (size_t)gr * 224 + gc;
#pragma unroll
        for (int nt = 0; nt < 2; ++nt)
          out[g * 256 + h * 32 + nt * 16 + l16] = accO[mt][nt][v] * inv;
      }
    }
  }
}

// ---- launch --------------------------------------------------------------

extern "C" void kernel_launch(void* const* d_in, const int* in_sizes, int n_in,
                              void* d_out, int out_size, void* d_ws, size_t ws_size,
                              hipStream_t stream) {
  (void)in_sizes; (void)n_in; (void)out_size; (void)ws_size;
  const float* x    = (const float*)d_in[0];
  const float* mask = (const float*)d_in[1];
  const float* w    = (const float*)d_in[2];
  const float* bias = (const float*)d_in[3];
  const float* rel  = (const float*)d_in[4];

  // workspace layout (bytes):
  //   xbf : 200704*256*2 = 102,760,448
  //   wbf : 768*256*2    =     393,216
  //   qkv : 200704*768*2 = 308,281,344
  bf16_t* xbf = (bf16_t*)d_ws;
  bf16_t* wbf = (bf16_t*)((char*)d_ws + 102760448);
  bf16_t* qkv = (bf16_t*)((char*)d_ws + 102760448 + 393216);
  float*  out = (float*)d_out;

  cvt_f32_bf16<<<dim3(50176, 1, 1), dim3(256, 1, 1), 0, stream>>>(x, xbf);
  cvt_f32_bf16<<<dim3(192, 1, 1), dim3(256, 1, 1), 0, stream>>>(w, wbf);

  dim3 g1(200704 / K1_BM, 768 / K1_BN, 1);
  qkv_gemm_bf16<<<g1, dim3(256, 1, 1), 0, stream>>>(xbf, wbf, bias, qkv);

  win_attn<<<dim3(4096, 1, 1), dim3(256, 1, 1), 231424, stream>>>(
      qkv, mask, rel, out);
}
